// VQLayer_49177375539794
// MI455X (gfx1250) — compile-verified, matrix operand order checked by reading, not run on hardware
//
#include <hip/hip_runtime.h>

typedef __bf16 v16bf __attribute__((ext_vector_type(16)));
typedef float  v8f   __attribute__((ext_vector_type(8)));

union ABFrag { uint4 u[2]; v16bf v; };

#define DDIM   256
#define KCODES 1024
#define NVEC   131072
#define NZ     (NVEC * DDIM)      // 33554432 floats of zq_st output
#define MTILE  16
#define NBLK   (NVEC / MTILE)     // 8192

__device__ __forceinline__ unsigned short f2bf(float f) {
  unsigned int u = __float_as_uint(f);
  u += 0x7fffu + ((u >> 16) & 1u);           // round-to-nearest-even
  return (unsigned short)(u >> 16);
}
__device__ __forceinline__ unsigned int pack2(float lo, float hi) {
  return (unsigned int)f2bf(lo) | ((unsigned int)f2bf(hi) << 16);
}

// ---------------- prologue: codebook fp32 -> bf16, |e_k|^2 ----------------
__global__ void __launch_bounds__(256) vq_prep(const float* __restrict__ cb,
                                               unsigned short* __restrict__ cbb,
                                               float* __restrict__ enorm) {
  const int k = blockIdx.x, t = threadIdx.x;
  const float v = cb[(size_t)k * DDIM + t];
  cbb[(size_t)k * DDIM + t] = f2bf(v);
  __shared__ float sm[256];
  sm[t] = v * v;
  __syncthreads();
  for (int o = 128; o > 0; o >>= 1) {
    if (t < o) sm[t] += sm[t + o];
    __syncthreads();
  }
  if (t == 0) enorm[k] = sm[0];
}

// ---------------- main: WMMA distance GEMM + argmin + gather --------------
__global__ void __launch_bounds__(256) vq_main(const float* __restrict__ z,
                                               const float* __restrict__ cb,
                                               const unsigned short* __restrict__ cbb,
                                               const float* __restrict__ enorm,
                                               float* __restrict__ out,
                                               float* __restrict__ partials) {
  __shared__ __align__(16) unsigned short zt[MTILE * DDIM];  // 8KB bf16 z tile
  __shared__ float red_s[8][16];
  __shared__ int   red_i[8][16];
  __shared__ int   fidx[16];
  __shared__ float rsum[256];

  const int t    = threadIdx.x;
  const int blk  = blockIdx.x;
  const int row0 = blk * MTILE;

  // Stage z tile to LDS as bf16: thread t handles row t/16, 16 cols
  {
    const int r = t >> 4, c0 = (t & 15) << 4;
    const float4* src = (const float4*)(z + (size_t)(row0 + r) * DDIM + c0);
    uint4* dst = (uint4*)(zt + r * DDIM + c0);
#pragma unroll
    for (int q = 0; q < 2; ++q) {
      float4 a = src[2 * q], b = src[2 * q + 1];
      uint4 o;
      o.x = pack2(a.x, a.y); o.y = pack2(a.z, a.w);
      o.z = pack2(b.x, b.y); o.w = pack2(b.z, b.w);
      dst[q] = o;
    }
  }
  __syncthreads();

  const int w  = t >> 5;   // wave 0..7
  const int L  = t & 31;
  const int m  = L & 15;   // A row / B column within tile
  const int hv = L >> 4;   // lane half selects K sub-span

  // A fragments for all 8 K-chunks (16 bf16 per lane per chunk), reused 8x
  ABFrag A[8];
#pragma unroll
  for (int kk = 0; kk < 8; ++kk) {
    const unsigned short* p = zt + m * DDIM + kk * 32 + hv * 8;
    A[kk].u[0] = *(const uint4*)p;
    A[kk].u[1] = *(const uint4*)(p + 16);
  }

  float best[8]; int bidx[8];
#pragma unroll
  for (int v = 0; v < 8; ++v) { best[v] = 3.4e38f; bidx[v] = 0x7fffffff; }

  for (int jt = 0; jt < 8; ++jt) {
    const int j    = jt * 8 + w;          // code tile 0..63
    const int code = j * 16 + m;          // this lane's code column
    const unsigned short* brow = cbb + (size_t)code * DDIM + hv * 8;
    v8f acc = {};
#pragma unroll
    for (int kk = 0; kk < 8; ++kk) {
      ABFrag B;
      B.u[0] = *(const uint4*)(brow + kk * 32);
      B.u[1] = *(const uint4*)(brow + kk * 32 + 16);
      acc = __builtin_amdgcn_wmma_f32_16x16x32_bf16(
          false, A[kk].v, false, B.v, (short)0, acc, false, false);
    }
    const float en = enorm[code];
#pragma unroll
    for (int v = 0; v < 8; ++v) {
      const float s = fmaf(-2.0f, acc[v], en);   // |e|^2 - 2 z.e
      if (s < best[v]) { best[v] = s; bidx[v] = code; }
    }
  }

  // reduce across the 16 lanes sharing each output row (xor stays in half)
#pragma unroll
  for (int v = 0; v < 8; ++v) {
    float s = best[v]; int i = bidx[v];
#pragma unroll
    for (int off = 1; off < 16; off <<= 1) {
      const float so = __shfl_xor(s, off, 32);
      const int   io = __shfl_xor(i, off, 32);
      if (so < s || (so == s && io < i)) { s = so; i = io; }
    }
    if (m == 0) { red_s[w][v + 8 * hv] = s; red_i[w][v + 8 * hv] = i; }
  }
  __syncthreads();

  // cross-wave argmin, emit indices
  if (t < 16) {
    float s = red_s[0][t]; int i = red_i[0][t];
#pragma unroll
    for (int ww = 1; ww < 8; ++ww) {
      const float so = red_s[ww][t]; const int io = red_i[ww][t];
      if (so < s || (so == s && io < i)) { s = so; i = io; }
    }
    fidx[t] = i;
    out[(size_t)NZ + row0 + t] = (float)i;   // min_ind output
  }
  __syncthreads();

  // gather fp32 codebook rows -> zq_st, accumulate loss partial
  {
    const int r = t >> 4, c0 = (t & 15) << 4;
    const int idx = fidx[r];
    const float4* ev = (const float4*)(cb + (size_t)idx * DDIM + c0);
    const float4* zv = (const float4*)(z + (size_t)(row0 + r) * DDIM + c0);
    float4* ov = (float4*)(out + (size_t)(row0 + r) * DDIM + c0);
    float d2 = 0.0f;
#pragma unroll
    for (int q = 0; q < 4; ++q) {
      const float4 e = ev[q], zz = zv[q];
      ov[q] = e;
      const float dx = e.x - zz.x, dy = e.y - zz.y;
      const float dz2 = e.z - zz.z, dw = e.w - zz.w;
      d2 += dx * dx + dy * dy + dz2 * dz2 + dw * dw;
    }
    rsum[t] = d2;
  }
  __syncthreads();
  for (int o = 128; o > 0; o >>= 1) {
    if (t < o) rsum[t] += rsum[t + o];
    __syncthreads();
  }
  if (t == 0) partials[blk] = rsum[0];
}

// ---------------- epilogue: deterministic loss reduction ------------------
__global__ void __launch_bounds__(256) vq_finalize(const float* __restrict__ partials,
                                                   float* __restrict__ loss_out) {
  __shared__ float sm[256];
  float s = 0.0f;
  for (int i = threadIdx.x; i < NBLK; i += 256) s += partials[i];
  sm[threadIdx.x] = s;
  __syncthreads();
  for (int o = 128; o > 0; o >>= 1) {
    if (threadIdx.x < o) sm[threadIdx.x] += sm[threadIdx.x + o];
    __syncthreads();
  }
  if (threadIdx.x == 0) loss_out[0] = 1.25f * sm[0] / (float)NZ;  // (1+beta)*MSE
}

extern "C" void kernel_launch(void* const* d_in, const int* in_sizes, int n_in,
                              void* d_out, int out_size, void* d_ws, size_t ws_size,
                              hipStream_t stream) {
  const float* z  = (const float*)d_in[0];   // [32,64,64,256] fp32
  const float* cb = (const float*)d_in[1];   // [1024,256] fp32
  float* out = (float*)d_out;                // zq_st | min_ind | loss

  unsigned short* cbb = (unsigned short*)d_ws;                       // 512KB bf16 codebook
  float* enorm    = (float*)((char*)d_ws + (size_t)KCODES * DDIM * 2); // 4KB |e|^2
  float* partials = enorm + KCODES;                                  // 32KB block partials

  vq_prep<<<KCODES, 256, 0, stream>>>(cb, cbb, enorm);
  vq_main<<<NBLK, 256, 0, stream>>>(z, cb, cbb, enorm, out, partials);
  vq_finalize<<<1, 256, 0, stream>>>(partials, out + (size_t)NZ + NVEC);
}
